// MDTA_86912958202231
// MI455X (gfx1250) — matrix operand order repaired and checked by reading, not compiled, not run on hardware
//
#include <hip/hip_runtime.h>
#include <stdint.h>

typedef __attribute__((ext_vector_type(16))) __bf16 v16bf;
typedef __attribute__((ext_vector_type(8)))  float  v8f;

#define B_     2
#define C_     128
#define H_     128
#define W_     128
#define HW_    (H_ * W_)
#define HEADS_ 8
#define D_     16

#define LDA_  40   // A tile K-stride (elements); 80B rows -> 16B aligned
#define LDN_  72   // B tile N-stride for [k][n] layout; 144B rows -> 16B aligned
#define LDNX_ 80   // conv halo tile N-stride; 160B rows -> 16B aligned

// ---- gfx1250 async global->LDS copies (ASYNCcnt path), with safe fallback --
#if defined(__has_builtin)
#if __has_builtin(__builtin_amdgcn_global_load_async_to_lds_b128) && \
    __has_builtin(__builtin_amdgcn_global_load_async_to_lds_b64)
#define MDTA_ASYNC 1
#endif
#if __has_builtin(__builtin_amdgcn_s_wait_asynccnt)
#define MDTA_HAVE_WAITA 1
#endif
#endif

#ifdef MDTA_ASYNC
typedef int mdta_v4i __attribute__((vector_size(16)));
typedef int mdta_v2i __attribute__((vector_size(8)));
typedef mdta_v4i __attribute__((address_space(1)))* mdta_g4p;
typedef mdta_v4i __attribute__((address_space(3)))* mdta_l4p;
typedef mdta_v2i __attribute__((address_space(1)))* mdta_g2p;
typedef mdta_v2i __attribute__((address_space(3)))* mdta_l2p;
#endif

__device__ __forceinline__ void cp16_g2l(void* dst_lds, const void* src_g) {
#ifdef MDTA_ASYNC
  __builtin_amdgcn_global_load_async_to_lds_b128(
      (mdta_g4p)src_g, (mdta_l4p)dst_lds, 0, 0);
#else
  *reinterpret_cast<uint4*>(dst_lds) = *reinterpret_cast<const uint4*>(src_g);
#endif
}
__device__ __forceinline__ void cp8_g2l(void* dst_lds, const void* src_g) {
#ifdef MDTA_ASYNC
  __builtin_amdgcn_global_load_async_to_lds_b64(
      (mdta_g2p)src_g, (mdta_l2p)dst_lds, 0, 0);
#else
  *reinterpret_cast<uint64_t*>(dst_lds) = *reinterpret_cast<const uint64_t*>(src_g);
#endif
}
__device__ __forceinline__ void wait_async() {
#ifdef MDTA_ASYNC
#ifdef MDTA_HAVE_WAITA
  __builtin_amdgcn_s_wait_asynccnt(0);
#else
  asm volatile("s_wait_asynccnt 0" ::: "memory");
#endif
#endif
}

__device__ __forceinline__ unsigned short f32_bf16(float f) {
  union { float f; uint32_t u; } c; c.f = f;
  uint32_t u = c.u;
  uint32_t r = (u + 0x7FFFu + ((u >> 16) & 1u)) >> 16;
  return (unsigned short)r;
}

// ---- WMMA bf16 fragment builders -------------------------------------------
// A (16x32): lane m = lane&15; lanes<16 hold K {0..7,16..23}, lanes>=16 K {8..15,24..31}
// element pairs are contiguous -> 4x 64-bit LDS loads per fragment.
__device__ __forceinline__ v16bf frag_a(const unsigned short* rowp, int lane) {
  const int kbase = (lane >> 4) << 3;
  union { v16bf v; uint64_t q[4]; } f;
  f.q[0] = *reinterpret_cast<const uint64_t*>(rowp + kbase);
  f.q[1] = *reinterpret_cast<const uint64_t*>(rowp + kbase + 4);
  f.q[2] = *reinterpret_cast<const uint64_t*>(rowp + 16 + kbase);
  f.q[3] = *reinterpret_cast<const uint64_t*>(rowp + 16 + kbase + 4);
  return f.v;
}
// B (32x16): lane n = lane&15; lanes<16 hold K {0..7,16..23}, lanes>=16 K {8..15,24..31}.
// From a [k][n] LDS tile: per-lane column walk (strided u16 reads).
__device__ __forceinline__ v16bf frag_b_strided(const unsigned short* colp,
                                                int ldn, int lane) {
  const unsigned short* p = colp + (((lane >> 4) << 3) * ldn);
  union { v16bf v; unsigned short s[16]; } f;
#pragma unroll
  for (int i = 0; i < 16; ++i) f.s[i] = p[(i < 8 ? i : i + 8) * ldn];
  return f.v;
}

// ---- 1. LayerNorm over last dim (W) per (b,c,h) row; f32 -> bf16 -----------
__global__ __launch_bounds__(256) void mdta_ln_kernel(
    const float* __restrict__ x, const float* __restrict__ gamma,
    const float* __restrict__ beta, unsigned short* __restrict__ xn) {
  int row  = blockIdx.x * 8 + (threadIdx.x >> 5);
  int lane = threadIdx.x & 31;
  const float4 v = reinterpret_cast<const float4*>(x + (size_t)row * W_)[lane];
  float s  = v.x + v.y + v.z + v.w;
  float sq = fmaf(v.x, v.x, fmaf(v.y, v.y, fmaf(v.z, v.z, v.w * v.w)));
#pragma unroll
  for (int off = 16; off; off >>= 1) {
    s  += __shfl_xor(s, off, 32);
    sq += __shfl_xor(sq, off, 32);
  }
  float mu  = s * (1.0f / W_);
  float var = sq * (1.0f / W_) - mu * mu;
  float rs  = rsqrtf(var + 1e-5f);
  const float4 g  = reinterpret_cast<const float4*>(gamma)[lane];
  const float4 bb = reinterpret_cast<const float4*>(beta)[lane];
  float r0 = (v.x - mu) * rs * g.x + bb.x;
  float r1 = (v.y - mu) * rs * g.y + bb.y;
  float r2 = (v.z - mu) * rs * g.z + bb.z;
  float r3 = (v.w - mu) * rs * g.w + bb.w;
  uint2 packed;
  packed.x = (uint32_t)f32_bf16(r0) | ((uint32_t)f32_bf16(r1) << 16);
  packed.y = (uint32_t)f32_bf16(r2) | ((uint32_t)f32_bf16(r3) << 16);
  reinterpret_cast<uint2*>(xn + (size_t)row * W_)[lane] = packed;
}

// ---- 2. weight conversion ---------------------------------------------------
__global__ void mdta_cvt_bf16(const float* __restrict__ in,
                              unsigned short* __restrict__ out, int n) {
  int i = blockIdx.x * 256 + threadIdx.x;
  if (i < n) out[i] = f32_bf16(in[i]);
}
// [o][c][3][3] f32 -> [tap][o][c] bf16
__global__ void mdta_cvt_w3(const float* __restrict__ in,
                            unsigned short* __restrict__ out) {
  int i = blockIdx.x * 256 + threadIdx.x;
  if (i < C_ * C_ * 9) {
    int oc = i / 9, tap = i % 9;
    out[tap * C_ * C_ + oc] = f32_bf16(in[i]);
  }
}

// ---- 3. 1x1 conv as GEMM: out[b,o,n] = sum_c Wb[o,c]*act[b,c,n]  (bf16 out)
// Block tile: M=128 (full output-channel stripe) x N=64.  Each wave: 4 M-tiles
// sharing one B fragment per K-step -> 4 WMMAs per (expensive) B fragment.
__global__ __launch_bounds__(256) void mdta_gemm1x1_kernel(
    const unsigned short* __restrict__ Wb, const unsigned short* __restrict__ act,
    unsigned short* __restrict__ out) {
  __shared__ __align__(16) unsigned short sA[128 * LDA_];  // [m][k]
  __shared__ __align__(16) unsigned short sB[32 * LDN_];   // [k][n]
  const int b = blockIdx.z, n0g = blockIdx.x * 64;
  const int tid = threadIdx.x, lane = tid & 31, wave = tid >> 5;
  const int wm = wave >> 2, wn = wave & 3;
  const unsigned short* Bp = act + (size_t)b * C_ * HW_;
  v8f acc[4] = {};
  for (int kt = 0; kt < C_; kt += 32) {
    // A tile: 128x32, 1024 async b64 chunks (4 per thread)
    for (int i = tid; i < 1024; i += 256) {
      int mm = i >> 3, k4 = (i & 7) * 4;
      cp8_g2l(&sA[mm * LDA_ + k4], &Wb[(size_t)mm * C_ + kt + k4]);
    }
    {  // B tile: 32x64, 256 async b128 chunks (one per thread)
      int kk = tid >> 3, nc = (tid & 7) * 8;
      cp16_g2l(&sB[kk * LDN_ + nc], &Bp[(size_t)(kt + kk) * HW_ + n0g + nc]);
    }
    wait_async();
    __syncthreads();
    v16bf fb = frag_b_strided(&sB[wn * 16 + (lane & 15)], LDN_, lane);
#pragma unroll
    for (int j = 0; j < 4; ++j) {
      v16bf fa = frag_a(&sA[(wm * 16 + 32 * j + (lane & 15)) * LDA_], lane);
      acc[j] = __builtin_amdgcn_wmma_f32_16x16x32_bf16(false, fa, false, fb,
                                                       (short)0, acc[j], false, false);
    }
    __syncthreads();
  }
  const int n = n0g + wn * 16 + (lane & 15);
  unsigned short* o = out + (size_t)b * C_ * HW_;
#pragma unroll
  for (int j = 0; j < 4; ++j) {
    const int mb = wm * 16 + 32 * j + ((lane >> 4) << 3);
#pragma unroll
    for (int g = 0; g < 8; ++g)
      o[(size_t)(mb + g) * HW_ + n] = f32_bf16(acc[j][g]);
  }
}

// ---- 4. 3x3 conv (pad 1): halo tile reused across 3 column taps x 4 M-tiles
__global__ __launch_bounds__(256) void mdta_conv3_kernel(
    const unsigned short* __restrict__ W3 /*[tap][o][c]*/,
    const unsigned short* __restrict__ in, unsigned short* __restrict__ out) {
  __shared__ __align__(16) unsigned short sA3[3][128 * LDA_];  // taps dh*3+{0,1,2}
  __shared__ __align__(16) unsigned short sBx[32 * LDNX_];     // [k][80] halo window
  const int b = blockIdx.z, n0g = blockIdx.x * 64;
  const int h = n0g >> 7, col0 = n0g & (W_ - 1);  // tiles never cross a row
  const int tid = threadIdx.x, lane = tid & 31, wave = tid >> 5;
  const int wm = wave >> 2, wn = wave & 3;
  v8f acc[4] = {};
  for (int dh = 0; dh < 3; ++dh) {
    const int hs = h + dh - 1;
    if ((unsigned)hs >= (unsigned)H_) continue;  // uniform skip: row is zero pad
    for (int kt = 0; kt < C_; kt += 32) {
      // A tiles for the 3 taps of this row: 3*128*32 elems = 3072 b64 chunks
      for (int i = tid; i < 3072; i += 256) {
        int t = i >> 10, r = i & 1023, mm = r >> 3, k4 = (r & 7) * 4;
        cp8_g2l(&sA3[t][mm * LDA_ + k4],
                &W3[(size_t)(dh * 3 + t) * C_ * C_ + (size_t)mm * C_ + kt + k4]);
      }
      // halo B tile: [32 k][80 cols] covering cols col0-8 .. col0+71
      for (int i = tid; i < 320; i += 256) {
        int kk = i / 10, cx = (i % 10) * 8;
        int wcol = col0 - 8 + cx;
        const unsigned short* src =
            in + ((size_t)b * C_ + kt + kk) * HW_ + (size_t)hs * W_ + wcol;
        unsigned short* dst = &sBx[kk * LDNX_ + cx];
        if (wcol >= 0 && wcol + 8 <= W_) {
          cp16_g2l(dst, src);
        } else {
#pragma unroll
          for (int e = 0; e < 8; ++e) {
            int c = wcol + e;
            dst[e] = ((unsigned)c < (unsigned)W_) ? src[e] : (unsigned short)0;
          }
        }
      }
      wait_async();
      __syncthreads();
#pragma unroll
      for (int dw = 0; dw < 3; ++dw) {
        v16bf fb = frag_b_strided(&sBx[wn * 16 + (lane & 15) + dw + 7], LDNX_, lane);
#pragma unroll
        for (int j = 0; j < 4; ++j) {
          v16bf fa = frag_a(&sA3[dw][(wm * 16 + 32 * j + (lane & 15)) * LDA_], lane);
          acc[j] = __builtin_amdgcn_wmma_f32_16x16x32_bf16(
              false, fa, false, fb, (short)0, acc[j], false, false);
        }
      }
      __syncthreads();
    }
  }
  const int n = n0g + wn * 16 + (lane & 15);
  unsigned short* o = out + (size_t)b * C_ * HW_;
#pragma unroll
  for (int j = 0; j < 4; ++j) {
    const int mb = wm * 16 + 32 * j + ((lane >> 4) << 3);
#pragma unroll
    for (int g = 0; g < 8; ++g)
      o[(size_t)(mb + g) * HW_ + n] = f32_bf16(acc[j][g]);
  }
}

// ---- 5. attention (16x16 per (b,h)) + softmax + Weff fusion ----------------
__global__ __launch_bounds__(256) void mdta_attn_kernel(
    const unsigned short* __restrict__ qc, const unsigned short* __restrict__ kc,
    const float* __restrict__ wo, const float* __restrict__ scale,
    unsigned short* __restrict__ weff) {
  const int bi = blockIdx.x, b = bi >> 3, h = bi & 7;
  const int tid = threadIdx.x, lane = tid & 31, wave = tid >> 5;
  const size_t chan = (size_t)(b * C_ + h * 16 + (lane & 15)) * HW_;
  const unsigned short* krow = kc + chan;  // A rows: d1
  const unsigned short* qrow = qc + chan;  // B cols: d2
  const int kbase = (lane >> 4) << 3;
  const int kbB   = (lane >> 4) << 3;
  v8f acc = {};
  for (int s = 0; s < HW_ / (8 * 32); ++s) {   // 64 K-steps of 32 per wave
    const int n0 = wave * (HW_ / 8) + s * 32;
    if ((s & 7) == 0) {  // stream-ahead prefetch (global_prefetch_b8)
      __builtin_prefetch(krow + n0 + 256, 0, 1);
      __builtin_prefetch(qrow + n0 + 256, 0, 1);
    }
    union { v16bf v; uint64_t q[4]; } fa;
    fa.q[0] = *reinterpret_cast<const uint64_t*>(krow + n0 + kbase);
    fa.q[1] = *reinterpret_cast<const uint64_t*>(krow + n0 + kbase + 4);
    fa.q[2] = *reinterpret_cast<const uint64_t*>(krow + n0 + 16 + kbase);
    fa.q[3] = *reinterpret_cast<const uint64_t*>(krow + n0 + 16 + kbase + 4);
    union { v16bf v; uint4 x[2]; } fb;
    fb.x[0] = *reinterpret_cast<const uint4*>(qrow + n0 + kbB);
    fb.x[1] = *reinterpret_cast<const uint4*>(qrow + n0 + 16 + kbB);
    acc = __builtin_amdgcn_wmma_f32_16x16x32_bf16(false, fa.v, false, fb.v,
                                                  (short)0, acc, false, false);
  }
  __shared__ float part[8][16][16];
  __shared__ float sm[16][17];
  __shared__ float sp[16][17];
#pragma unroll
  for (int g = 0; g < 8; ++g)
    part[wave][g + ((lane >> 4) << 3)][lane & 15] = acc[g];
  __syncthreads();
  {  // cross-wave reduce + scale
    int m = tid >> 4, n = tid & 15;
    float s = 0.f;
#pragma unroll
    for (int w = 0; w < 8; ++w) s += part[w][m][n];
    sm[m][n] = s / scale[0];
  }
  __syncthreads();
  if (tid < 16) {  // softmax over last axis (tiny: 16x16)
    float mx = -3.0e38f;
#pragma unroll
    for (int n = 0; n < 16; ++n) mx = fmaxf(mx, sm[tid][n]);
    float ssum = 0.f;
#pragma unroll
    for (int n = 0; n < 16; ++n) {
      float e = __expf(sm[tid][n] - mx);
      sp[tid][n] = e;
      ssum += e;
    }
    float inv = 1.0f / ssum;
#pragma unroll
    for (int n = 0; n < 16; ++n) sp[tid][n] *= inv;
  }
  __syncthreads();
  // Weff[b][o][h*16+d1] = sum_d2 wo[o][h*16+d2] * probs[d1][d2]
  for (int i = tid; i < C_ * 16; i += 256) {
    int o = i >> 4, d1 = i & 15;
    float s = 0.f;
#pragma unroll
    for (int d2 = 0; d2 < 16; ++d2)
      s += wo[o * C_ + h * 16 + d2] * sp[d1][d2];
    weff[((size_t)b * C_ + o) * C_ + h * 16 + d1] = f32_bf16(s);
  }
}

// ---- 6. final per-batch GEMM: out = Weff[b] @ v + residual  (f32 out) ------
__global__ __launch_bounds__(256) void mdta_final_kernel(
    const unsigned short* __restrict__ weff, const unsigned short* __restrict__ vc,
    const float* __restrict__ xres, float* __restrict__ out) {
  __shared__ __align__(16) unsigned short sA[128 * LDA_];
  __shared__ __align__(16) unsigned short sB[32 * LDN_];
  const int b = blockIdx.z, n0g = blockIdx.x * 64;
  const int tid = threadIdx.x, lane = tid & 31, wave = tid >> 5;
  const int wm = wave >> 2, wn = wave & 3;
  const unsigned short* A  = weff + (size_t)b * C_ * C_;
  const unsigned short* Bp = vc + (size_t)b * C_ * HW_;
  v8f acc[4] = {};
  for (int kt = 0; kt < C_; kt += 32) {
    for (int i = tid; i < 1024; i += 256) {
      int mm = i >> 3, k4 = (i & 7) * 4;
      cp8_g2l(&sA[mm * LDA_ + k4], &A[(size_t)mm * C_ + kt + k4]);
    }
    {
      int kk = tid >> 3, nc = (tid & 7) * 8;
      cp16_g2l(&sB[kk * LDN_ + nc], &Bp[(size_t)(kt + kk) * HW_ + n0g + nc]);
    }
    wait_async();
    __syncthreads();
    v16bf fb = frag_b_strided(&sB[wn * 16 + (lane & 15)], LDN_, lane);
#pragma unroll
    for (int j = 0; j < 4; ++j) {
      v16bf fa = frag_a(&sA[(wm * 16 + 32 * j + (lane & 15)) * LDA_], lane);
      acc[j] = __builtin_amdgcn_wmma_f32_16x16x32_bf16(false, fa, false, fb,
                                                       (short)0, acc[j], false, false);
    }
    __syncthreads();
  }
  const int n = n0g + wn * 16 + (lane & 15);
#pragma unroll
  for (int j = 0; j < 4; ++j) {
    const int mb = wm * 16 + 32 * j + ((lane >> 4) << 3);
#pragma unroll
    for (int g = 0; g < 8; ++g) {
      size_t idx = ((size_t)b * C_ + mb + g) * HW_ + n;
      out[idx] = acc[j][g] + xres[idx];
    }
  }
}

// ---- host: workspace carving + launches ------------------------------------
extern "C" void kernel_launch(void* const* d_in, const int* in_sizes, int n_in,
                              void* d_out, int out_size, void* d_ws, size_t ws_size,
                              hipStream_t stream) {
  const float* x     = (const float*)d_in[0];
  const float* gamma = (const float*)d_in[1];
  const float* beta  = (const float*)d_in[2];
  const float* scale = (const float*)d_in[3];
  const float* wq1 = (const float*)d_in[4];
  const float* wq2 = (const float*)d_in[5];
  const float* wk1 = (const float*)d_in[6];
  const float* wk2 = (const float*)d_in[7];
  const float* wv1 = (const float*)d_in[8];
  const float* wv2 = (const float*)d_in[9];
  const float* wo  = (const float*)d_in[10];
  float* out = (float*)d_out;

  const size_t TEN = (size_t)B_ * C_ * HW_;
  char* ws = (char*)d_ws;
  size_t off = 0;
  auto carve = [&](size_t bytes) {
    char* p = ws + off;
    off = (off + bytes + 255) & ~(size_t)255;
    return p;
  };
  unsigned short* xn  = (unsigned short*)carve(TEN * 2);
  unsigned short* q1  = (unsigned short*)carve(TEN * 2);
  unsigned short* k1  = (unsigned short*)carve(TEN * 2);
  unsigned short* v1  = (unsigned short*)carve(TEN * 2);
  unsigned short* qc  = (unsigned short*)carve(TEN * 2);
  unsigned short* kc  = (unsigned short*)carve(TEN * 2);
  unsigned short* vc  = (unsigned short*)carve(TEN * 2);
  unsigned short* wq1b = (unsigned short*)carve((size_t)C_ * C_ * 2);
  unsigned short* wk1b = (unsigned short*)carve((size_t)C_ * C_ * 2);
  unsigned short* wv1b = (unsigned short*)carve((size_t)C_ * C_ * 2);
  unsigned short* w3q  = (unsigned short*)carve((size_t)C_ * C_ * 9 * 2);
  unsigned short* w3k  = (unsigned short*)carve((size_t)C_ * C_ * 9 * 2);
  unsigned short* w3v  = (unsigned short*)carve((size_t)C_ * C_ * 9 * 2);
  unsigned short* weff = (unsigned short*)carve((size_t)B_ * C_ * C_ * 2);
  (void)ws_size; (void)in_sizes; (void)n_in; (void)out_size;

  mdta_ln_kernel<<<(B_ * C_ * H_) / 8, 256, 0, stream>>>(x, gamma, beta, xn);
  mdta_cvt_bf16<<<(C_ * C_ + 255) / 256, 256, 0, stream>>>(wq1, wq1b, C_ * C_);
  mdta_cvt_bf16<<<(C_ * C_ + 255) / 256, 256, 0, stream>>>(wk1, wk1b, C_ * C_);
  mdta_cvt_bf16<<<(C_ * C_ + 255) / 256, 256, 0, stream>>>(wv1, wv1b, C_ * C_);
  mdta_cvt_w3<<<(C_ * C_ * 9 + 255) / 256, 256, 0, stream>>>(wq2, w3q);
  mdta_cvt_w3<<<(C_ * C_ * 9 + 255) / 256, 256, 0, stream>>>(wk2, w3k);
  mdta_cvt_w3<<<(C_ * C_ * 9 + 255) / 256, 256, 0, stream>>>(wv2, w3v);

  dim3 gg(HW_ / 64, 1, B_);
  mdta_gemm1x1_kernel<<<gg, 256, 0, stream>>>(wq1b, xn, q1);
  mdta_gemm1x1_kernel<<<gg, 256, 0, stream>>>(wk1b, xn, k1);
  mdta_gemm1x1_kernel<<<gg, 256, 0, stream>>>(wv1b, xn, v1);
  mdta_conv3_kernel<<<gg, 256, 0, stream>>>(w3q, q1, qc);
  mdta_conv3_kernel<<<gg, 256, 0, stream>>>(w3k, k1, kc);
  mdta_conv3_kernel<<<gg, 256, 0, stream>>>(w3v, v1, vc);
  mdta_attn_kernel<<<B_ * HEADS_, 256, 0, stream>>>(qc, kc, wo, scale, weff);
  mdta_final_kernel<<<gg, 256, 0, stream>>>(weff, vc, x, out);
}
